// PointAttention_42417097015910
// MI455X (gfx1250) — matrix operand order repaired and checked by reading, not compiled, not run on hardware
//
#include <hip/hip_runtime.h>
#include <hip/hip_bf16.h>
#include <math.h>

#define NPTS 8192
#define BATCH 4
#define DMODEL 128
#define KNN 8

typedef __attribute__((ext_vector_type(16))) _Float16 v16h;
typedef __attribute__((ext_vector_type(8)))  _Float16 v8h;
typedef __attribute__((ext_vector_type(8)))  float    v8f;

__device__ __forceinline__ v8f wmma_f16(v16h a, v16h b, v8f c) {
  // D = A(16x32 f16) x B(32x16 f16) + C(16x16 f32)
  return __builtin_amdgcn_wmma_f32_16x16x32_f16(false, a, false, b, (short)0, c, false, false);
}

// A fragment (16x32) from LDS row-major f16 [rows][ldk], tile at (row0, ks*32).
// Per ISA 7.12.2: lanes 0-15 hold M=0..15 / K=(0..7,16..23); lanes 16-31 M=0..15 / K=(8..15,24..31).
__device__ __forceinline__ v16h load_a_lds(const _Float16* base, int row0, int ldk, int ks, int lane) {
  int m  = lane & 15;
  int kb = ks * 32 + ((lane >> 4) << 3);
  const _Float16* p = base + (row0 + m) * ldk + kb;
  v8h lo = *(const v8h*)(p);        // K = kb .. kb+7
  v8h hi = *(const v8h*)(p + 16);   // K = kb+16 .. kb+23
  v16h a;
#pragma unroll
  for (int i = 0; i < 8; ++i) { a[i] = lo[i]; a[i + 8] = hi[i]; }
  return a;
}

// B fragment (32x16) from pre-packed weights: one contiguous 32B load per lane.
__device__ __forceinline__ v16h load_b_pk(const _Float16* pk, int kt, int nt, int lane) {
  return *(const v16h*)(pk + (((kt << 3) + nt) * 32 + lane) * 16);
}

// Pack six 128x128 f32 weights into B-fragment layout:
// lane l of tile (kt,nt): n = nt*16 + (l&15), K = kt*32 + (l>>4)*8 + (i&7) + (i>>3)*16, i=0..15.
__global__ void pack_weights_kernel(const float* w0, const float* w1, const float* w2,
                                    const float* w3, const float* w4, const float* w5,
                                    _Float16* __restrict__ packed) {
  int blk  = blockIdx.x;
  int wi   = blk >> 5;
  int tile = blk & 31;          // tile = kt*8 + nt
  int kt = tile >> 3, nt = tile & 7;
  const float* W = wi == 0 ? w0 : wi == 1 ? w1 : wi == 2 ? w2 : wi == 3 ? w3 : wi == 4 ? w4 : w5;
  int lane = threadIdx.x;
  int nn = nt * 16 + (lane & 15);
  int kb = kt * 32 + ((lane >> 4) << 3);
  _Float16* dst = packed + wi * 16384 + (tile * 32 + lane) * 16;
#pragma unroll
  for (int i = 0; i < 16; ++i) dst[i] = (_Float16)W[(kb + (i & 7) + ((i >> 3) << 4)) * DMODEL + nn];
}

// Brute-force KNN: one thread per query point, LDS-tiled candidates, register top-8.
__global__ __launch_bounds__(256) void knn_kernel(const float* __restrict__ pos,
                                                  int* __restrict__ idx) {
  __shared__ float sp[256 * 3];
  int b = blockIdx.y;
  int n = blockIdx.x * 256 + threadIdx.x;
  const float* pb = pos + (size_t)b * NPTS * 3;
  float px = pb[n * 3 + 0], py = pb[n * 3 + 1], pz = pb[n * 3 + 2];
  float bd[8]; int bi[8];
#pragma unroll
  for (int i = 0; i < 8; ++i) { bd[i] = 3.4e38f; bi[i] = 0; }
  for (int t0 = 0; t0 < NPTS; t0 += 256) {
    __syncthreads();
#pragma unroll
    for (int u = 0; u < 3; ++u) {
      int fl = u * 256 + threadIdx.x;
      sp[fl] = pb[t0 * 3 + fl];
    }
    __syncthreads();
    for (int j = 0; j < 256; ++j) {
      float dx = px - sp[j * 3 + 0], dy = py - sp[j * 3 + 1], dz = pz - sp[j * 3 + 2];
      float d2 = dx * dx + dy * dy + dz * dz;
      if (d2 < bd[7]) {
        bd[7] = d2; bi[7] = t0 + j;
#pragma unroll
        for (int p = 7; p > 0; --p) {   // one bubble pass restores sorted order
          if (bd[p] < bd[p - 1]) {
            float td = bd[p]; bd[p] = bd[p - 1]; bd[p - 1] = td;
            int   ti = bi[p]; bi[p] = bi[p - 1]; bi[p - 1] = ti;
          }
        }
      }
    }
  }
#pragma unroll
  for (int i = 0; i < 8; ++i) idx[((size_t)b * NPTS + n) * KNN + i] = bi[i];
}

// q/k/v projections: rows are query[N,B,D] flattened (r = n*B + b); outputs in [B,N,D].
__global__ __launch_bounds__(256) void qkv_kernel(const float* __restrict__ query,
    const _Float16* __restrict__ pw,
    const float* __restrict__ bq, const float* __restrict__ bk, const float* __restrict__ bv,
    float* __restrict__ qf, float* __restrict__ kf, float* __restrict__ vf) {
  __shared__ __align__(16) _Float16 s_a[16 * 128];
  int t = threadIdx.x;
  int gr0 = blockIdx.x * 16;
#pragma unroll
  for (int u = 0; u < 8; ++u) {
    int fl = u * 256 + t;
    s_a[fl] = (_Float16)query[(size_t)gr0 * DMODEL + fl];
  }
  __syncthreads();
  int w = t >> 5, lane = t & 31;
  float* outs[3] = {qf, kf, vf};
  const float* bias[3] = {bq, bk, bv};
#pragma unroll
  for (int wi = 0; wi < 3; ++wi) {
    v8f acc = {};
#pragma unroll
    for (int ks = 0; ks < 4; ++ks)
      acc = wmma_f16(load_a_lds(s_a, 0, 128, ks, lane), load_b_pk(pw + wi * 16384, ks, w, lane), acc);
    int col = w * 16 + (lane & 15);
    float bb = bias[wi][col];
#pragma unroll
    for (int r = 0; r < 8; ++r) {
      int m = r + ((lane >> 4) << 3);       // C layout: M = r + 8*(lane>=16), N = lane&15
      int gr = gr0 + m;
      int n = gr >> 2, bidx = gr & 3;       // r = n*B + b, B = 4
      outs[wi][((size_t)bidx * NPTS + n) * DMODEL + col] = acc[r] + bb;
    }
  }
}

// Fused: pe MLP (gelu + WMMA), attn linear (WMMA), feature softmax, K-reduction,
// output projection (WMMA) + residual. One block per (16 points, batch).
__global__ __launch_bounds__(256) void fused_attn_kernel(
    const float* __restrict__ query, const float* __restrict__ pos,
    const float* __restrict__ qf, const float* __restrict__ kf, const float* __restrict__ vf,
    const int* __restrict__ knn_idx,
    const _Float16* __restrict__ pWp2, const _Float16* __restrict__ pWg, const _Float16* __restrict__ pWo,
    const float* __restrict__ Wp1, const float* __restrict__ bp1,
    const float* __restrict__ bp2, const float* __restrict__ bg, const float* __restrict__ bo,
    float* __restrict__ out) {
  __shared__ int   s_nbr[128];
  __shared__ float s_rel[128 * 3];
  __shared__ float s_q[16 * 128];
  __shared__ __align__(16) _Float16 s_x[128 * 128];   // h(gelu) then attn_in
  __shared__ __align__(16) _Float16 s_pe[128 * 128];
  __shared__ float s_attn[128 * 128];
  __shared__ __align__(16) _Float16 s_res[16 * 128];

  int t = threadIdx.x;
  int b = blockIdx.y;
  int n0 = blockIdx.x * 16;
  const float* posb = pos + (size_t)b * NPTS * 3;

  if (t < 128) {
    int i = t >> 3;
    int j = knn_idx[((size_t)b * NPTS + n0 + i) * KNN + (t & 7)];
    s_nbr[t] = j;
#pragma unroll
    for (int c = 0; c < 3; ++c)
      s_rel[t * 3 + c] = posb[(n0 + i) * 3 + c] - posb[j * 3 + c];
  }
#pragma unroll
  for (int u = 0; u < 8; ++u) {
    int fl = u * 256 + t;
    s_q[fl] = qf[((size_t)b * NPTS + n0) * DMODEL + fl];
  }
  __syncthreads();

  // pe layer 1 (3->128) + exact gelu, into s_x as f16
#pragma unroll 4
  for (int u = 0; u < 64; ++u) {
    int fl = u * 256 + t;
    int rr = fl >> 7, d = fl & 127;
    float x = s_rel[rr * 3 + 0] * Wp1[d] + s_rel[rr * 3 + 1] * Wp1[128 + d]
            + s_rel[rr * 3 + 2] * Wp1[256 + d] + bp1[d];
    float g = 0.5f * x * (1.0f + erff(x * 0.70710678118654752f));
    s_x[fl] = (_Float16)g;
  }
  __syncthreads();

  int w = t >> 5, lane = t & 31;
  int rt = w;  // each wave owns rows [rt*16, rt*16+16)

  // pe = gelu(h) @ Wp2 + bp2  (wave-local rows, no barrier needed)
#pragma unroll
  for (int nt = 0; nt < 8; ++nt) {
    v8f acc = {};
#pragma unroll
    for (int ks = 0; ks < 4; ++ks)
      acc = wmma_f16(load_a_lds(s_x, rt * 16, 128, ks, lane), load_b_pk(pWp2, ks, nt, lane), acc);
    int col = nt * 16 + (lane & 15);
    float bb = bp2[col];
#pragma unroll
    for (int r = 0; r < 8; ++r) {
      int rr = rt * 16 + r + ((lane >> 4) << 3);
      s_pe[rr * 128 + col] = (_Float16)(acc[r] + bb);
    }
  }
  // attn_in = q - k_knn + pe  (overwrite own rows of s_x; k gathers hit L2)
#pragma unroll
  for (int nt = 0; nt < 8; ++nt) {
    int col = nt * 16 + (lane & 15);
#pragma unroll
    for (int r = 0; r < 8; ++r) {
      int rr = rt * 16 + r + ((lane >> 4) << 3);
      int i = rr >> 3;
      int j = s_nbr[rr];
      float ai = s_q[i * 128 + col] - kf[((size_t)b * NPTS + j) * DMODEL + col]
               + (float)s_pe[rr * 128 + col];
      s_x[rr * 128 + col] = (_Float16)ai;
    }
  }
  // attn = attn_in @ Wg + bg
#pragma unroll
  for (int nt = 0; nt < 8; ++nt) {
    v8f acc = {};
#pragma unroll
    for (int ks = 0; ks < 4; ++ks)
      acc = wmma_f16(load_a_lds(s_x, rt * 16, 128, ks, lane), load_b_pk(pWg, ks, nt, lane), acc);
    int col = nt * 16 + (lane & 15);
    float bb = bg[col];
#pragma unroll
    for (int r = 0; r < 8; ++r) {
      int rr = rt * 16 + r + ((lane >> 4) << 3);
      s_attn[rr * 128 + col] = acc[r] + bb;
    }
  }
  __syncthreads();

  // softmax over the FEATURE dim (as in reference), scaled by 1/sqrt(D)
  if (t < 128) {
    const float sc = 0.088388347648318447f; // 1/sqrt(128)
    float* row = s_attn + t * 128;
    float mx = -3.4e38f;
    for (int c = 0; c < 128; ++c) mx = fmaxf(mx, row[c] * sc);
    float sum = 0.f;
    for (int c = 0; c < 128; ++c) { float e = __expf(row[c] * sc - mx); row[c] = e; sum += e; }
    float inv = 1.0f / sum;
    for (int c = 0; c < 128; ++c) row[c] *= inv;
  }
  __syncthreads();

  // res[i][d] = sum_k attn[i,k,d] * (v_knn[i,k,d] + pe[i,k,d])
#pragma unroll
  for (int u = 0; u < 8; ++u) {
    int fl = u * 256 + t;
    int i = fl >> 7, d = fl & 127;
    float acc = 0.f;
#pragma unroll
    for (int kk = 0; kk < 8; ++kk) {
      int rr = i * 8 + kk;
      acc += s_attn[rr * 128 + d] *
             (vf[((size_t)b * NPTS + s_nbr[rr]) * DMODEL + d] + (float)s_pe[rr * 128 + d]);
    }
    s_res[i * 128 + d] = (_Float16)acc;
  }
  __syncthreads();

  // out = res @ Wo + bo + residual (out layout [N,B,D])
  {
    int nt = w;
    v8f acc = {};
#pragma unroll
    for (int ks = 0; ks < 4; ++ks)
      acc = wmma_f16(load_a_lds(s_res, 0, 128, ks, lane), load_b_pk(pWo, ks, nt, lane), acc);
    int col = nt * 16 + (lane & 15);
    float bb = bo[col];
#pragma unroll
    for (int r = 0; r < 8; ++r) {
      int m = r + ((lane >> 4) << 3);
      size_t gi = ((size_t)(n0 + m) * BATCH + b) * DMODEL + col;
      out[gi] = acc[r] + bb + query[gi];
    }
  }
}

extern "C" void kernel_launch(void* const* d_in, const int* in_sizes, int n_in,
                              void* d_out, int out_size, void* d_ws, size_t ws_size,
                              hipStream_t stream) {
  const float* query = (const float*)d_in[0];
  const float* pos   = (const float*)d_in[1];
  const float* Wq  = (const float*)d_in[2];  const float* bq  = (const float*)d_in[3];
  const float* Wk  = (const float*)d_in[4];  const float* bk  = (const float*)d_in[5];
  const float* Wv  = (const float*)d_in[6];  const float* bv  = (const float*)d_in[7];
  const float* Wp1 = (const float*)d_in[8];  const float* bp1 = (const float*)d_in[9];
  const float* Wp2 = (const float*)d_in[10]; const float* bp2 = (const float*)d_in[11];
  const float* Wg  = (const float*)d_in[12]; const float* bg  = (const float*)d_in[13];
  const float* Wo  = (const float*)d_in[14]; const float* bo  = (const float*)d_in[15];
  float* out = (float*)d_out;

  char* ws = (char*)d_ws;
  _Float16* pw = (_Float16*)ws;                      // 6 packed weights, 32KB each
  size_t off = (size_t)6 * 16384 * sizeof(_Float16); // 196608 B
  float* qf = (float*)(ws + off); off += (size_t)NPTS * BATCH * DMODEL * 4;
  float* kf = (float*)(ws + off); off += (size_t)NPTS * BATCH * DMODEL * 4;
  float* vf = (float*)(ws + off); off += (size_t)NPTS * BATCH * DMODEL * 4;
  int* knn  = (int*)(ws + off);                      // [B,N,K]

  pack_weights_kernel<<<6 * 32, 32, 0, stream>>>(Wq, Wk, Wv, Wp2, Wg, Wo, pw);
  knn_kernel<<<dim3(NPTS / 256, BATCH), 256, 0, stream>>>(pos, knn);
  qkv_kernel<<<(NPTS * BATCH) / 16, 256, 0, stream>>>(query, pw, bq, bk, bv, qf, kf, vf);
  fused_attn_kernel<<<dim3(NPTS / 16, BATCH), 256, 0, stream>>>(
      query, pos, qf, kf, vf, knn,
      pw + 3 * 16384, pw + 4 * 16384, pw + 5 * 16384,
      Wp1, bp1, bp2, bg, bo, out);
}
